// GatedResidueNorm_84559316123891
// MI455X (gfx1250) — compile-verified, matrix operand order checked by reading, not run on hardware
//
#include <hip/hip_runtime.h>

// ---------------------------------------------------------------------------
// GatedResidueNorm for MI455X (gfx1250, wave32, WMMA)
//   d = 256, 3d = 768, gate GEMM done in bf16 WMMA (f32 accumulate)
// ---------------------------------------------------------------------------

typedef __attribute__((ext_vector_type(16))) __bf16           v16bf;
typedef __attribute__((ext_vector_type(8)))  float            v8f;
typedef __attribute__((ext_vector_type(4)))  float            fx4;
typedef __attribute__((ext_vector_type(8)))  unsigned short   usx8;
typedef __attribute__((ext_vector_type(4)))  unsigned short   usx4;

struct BfFrag { usx8 lo, hi; };   // 32 bytes == v16bf

static __device__ __forceinline__ unsigned short f2bf(float f) {
    // round-to-nearest-even f32 -> bf16
    unsigned u = __float_as_uint(f);
    u += 0x7FFFu + ((u >> 16) & 1u);
    return (unsigned short)(u >> 16);
}

constexpr int   DDIM   = 256;       // d
constexpr int   KDIM   = 768;       // 3d
constexpr int   KTILES = KDIM / 32; // 24 k-tiles of 32 (bf16 wmma K)
constexpr int   NTILES = DDIM / 16; // 16 n-tiles of 16
constexpr float EPS    = 1e-5f;

// ---------------------------------------------------------------------------
// Kernel 1: repack W_gate [768,256] f32 -> bf16 in WMMA B-fragment order.
// Fragment index g enumerates (nt, kt, lane) with lane fastest; each lane
// stores its 16 bf16 elements contiguously (32B), so the GEMM loop does two
// aligned b128 loads per fragment. K mapping per the ISA 16-bit layout:
//   base = (lane>>4)*8 ; j<8 -> K = base+j ; j>=8 -> K = 16+base+(j-8)
// ---------------------------------------------------------------------------
__global__ __launch_bounds__(256)
void wconvert_kernel(const float* __restrict__ W, unsigned short* __restrict__ wbf) {
    int g = blockIdx.x * 256 + threadIdx.x;
    if (g >= NTILES * KTILES * 32) return;
    int lane = g & 31;
    int kt   = (g >> 5) % KTILES;
    int nt   = (g >> 5) / KTILES;
    int base = (lane >> 4) * 8;
    int n    = nt * 16 + (lane & 15);
    size_t off = (size_t)g * 16;
#pragma unroll
    for (int j = 0; j < 16; ++j) {
        int k = kt * 32 + ((j < 8) ? (base + j) : (16 + base + (j - 8)));
        wbf[off + j] = f2bf(W[(size_t)k * DDIM + n]);
    }
}

// ---------------------------------------------------------------------------
// Kernel 2: LayerNorm of every row of G_t into out (base pass).
// One wave32 per row; lane holds 8 contiguous floats (two float4 loads).
// ---------------------------------------------------------------------------
__global__ __launch_bounds__(256)
void lnall_kernel(const float* __restrict__ Gt,
                  const float* __restrict__ gamma,
                  const float* __restrict__ beta,
                  float* __restrict__ out, int N) {
    int wave = threadIdx.x >> 5;
    int lane = threadIdx.x & 31;
    int row  = blockIdx.x * 8 + wave;
    if (row >= N) return;

    const float* x = Gt + (size_t)row * DDIM;
    int c0 = lane * 8;
    fx4 a = *(const fx4*)(x + c0);
    fx4 b = *(const fx4*)(x + c0 + 4);

    float s = 0.f, ss = 0.f;
#pragma unroll
    for (int j = 0; j < 4; ++j) { s += a[j]; ss += a[j] * a[j]; }
#pragma unroll
    for (int j = 0; j < 4; ++j) { s += b[j]; ss += b[j] * b[j]; }
#pragma unroll
    for (int o = 16; o > 0; o >>= 1) {
        s  += __shfl_xor(s,  o);
        ss += __shfl_xor(ss, o);
    }
    float mu  = s * (1.f / DDIM);
    float var = ss * (1.f / DDIM) - mu * mu;
    float rs  = rsqrtf(var + EPS);

    fx4 g0 = *(const fx4*)(gamma + c0);
    fx4 g1 = *(const fx4*)(gamma + c0 + 4);
    fx4 b0 = *(const fx4*)(beta + c0);
    fx4 b1 = *(const fx4*)(beta + c0 + 4);
    fx4 y0, y1;
#pragma unroll
    for (int j = 0; j < 4; ++j) y0[j] = (a[j] - mu) * rs * g0[j] + b0[j];
#pragma unroll
    for (int j = 0; j < 4; ++j) y1[j] = (b[j] - mu) * rs * g1[j] + b1[j];

    float* o = out + (size_t)row * DDIM + c0;
    *(fx4*)o       = y0;
    *(fx4*)(o + 4) = y1;
}

// ---------------------------------------------------------------------------
// Kernel 3: gate GEMM + sigmoid + residual + LayerNorm + scatter.
// One block = 16 rows of M. 8 waves; wave w owns n-tiles 2w, 2w+1.
// LDS: 16x768 bf16 A-tile (24 KB), later aliased as 16x256 f32 result tile.
// ---------------------------------------------------------------------------
__global__ __launch_bounds__(256)
void gate_kernel(const float* __restrict__ Gt,
                 const float* __restrict__ Gh,
                 const float* __restrict__ q,
                 const int*   __restrict__ hmap,
                 const unsigned short* __restrict__ wbf,
                 const float* __restrict__ gamma,
                 const float* __restrict__ beta,
                 float* __restrict__ out, int M) {
    __shared__ __align__(16) unsigned char smem[16 * KDIM * 2]; // 24576 B
    unsigned short* sA   = (unsigned short*)smem;   // [16][768] bf16
    float*          sOut = (float*)smem;            // [16][256] f32 (aliased)
    __shared__ int sMap[16];   // gathered global G_t row per local row
    __shared__ int sRow[16];   // clamped global M-row per local row

    int tid = threadIdx.x;
    int m0  = blockIdx.x * 16;

    if (tid < 16) {
        int m = m0 + tid;
        if (m > M - 1) m = M - 1;          // safe tail (dup writes identical)
        sRow[tid] = m;
        sMap[tid] = hmap[m];
    }
    __syncthreads();

    // ---- phase 1: build bf16 gate_in tile [16][768] in LDS ----
    for (int idx = tid; idx < 16 * (KDIM / 4); idx += 256) {
        int r   = idx / (KDIM / 4);
        int col = (idx - r * (KDIM / 4)) * 4;
        fx4 v;
        if (col < 256)
            v = *(const fx4*)(Gt + (size_t)sMap[r] * DDIM + col);
        else if (col < 512)
            v = *(const fx4*)(Gh + (size_t)sRow[r] * DDIM + (col - 256));
        else
            v = *(const fx4*)(q  + (size_t)sRow[r] * DDIM + (col - 512));
        usx4 h;
#pragma unroll
        for (int j = 0; j < 4; ++j) h[j] = f2bf(v[j]);
        *(usx4*)(sA + r * KDIM + col) = h;
    }
    __syncthreads();

    // ---- phase 2: WMMA GEMM, K = 768 in 24 tiles of 32 ----
    int wave = tid >> 5;
    int lane = tid & 31;
    int nt0  = wave * 2;
    int nt1  = nt0 + 1;

    v8f c0, c1;
#pragma unroll
    for (int i = 0; i < 8; ++i) { c0[i] = 0.f; c1[i] = 0.f; }

    int rowA  = lane & 15;
    int kbase = (lane >> 4) * 8;
    const usx8* bp = (const usx8*)wbf;

    for (int kt = 0; kt < KTILES; ++kt) {
        BfFrag fa;
        fa.lo = *(const usx8*)(sA + rowA * KDIM + kt * 32 + kbase);
        fa.hi = *(const usx8*)(sA + rowA * KDIM + kt * 32 + kbase + 16);
        v16bf a = __builtin_bit_cast(v16bf, fa);

        size_t f0 = ((size_t)(nt0 * KTILES + kt) * 32 + lane) * 2;
        size_t f1 = ((size_t)(nt1 * KTILES + kt) * 32 + lane) * 2;
        BfFrag fb0, fb1;
        fb0.lo = bp[f0];     fb0.hi = bp[f0 + 1];
        fb1.lo = bp[f1];     fb1.hi = bp[f1 + 1];
        v16bf b0 = __builtin_bit_cast(v16bf, fb0);
        v16bf b1 = __builtin_bit_cast(v16bf, fb1);

        c0 = __builtin_amdgcn_wmma_f32_16x16x32_bf16(
                 false, a, false, b0, (short)0, c0, false, false);
        c1 = __builtin_amdgcn_wmma_f32_16x16x32_bf16(
                 false, a, false, b1, (short)0, c1, false, false);
    }
    __syncthreads();   // all waves done reading sA; safe to alias as sOut

    // ---- phase 3: sigmoid gate + residual, stage rows in LDS ----
    // C/D layout: VGPR i, lanes 0-15 -> M=i, lanes 16-31 -> M=i+8; N = lane&15.
    int mhalf = (lane >> 4) * 8;
    int nlo   = lane & 15;
#pragma unroll
    for (int t = 0; t < 2; ++t) {
        v8f& c  = t ? c1 : c0;
        int col = (t ? nt1 : nt0) * 16 + nlo;
#pragma unroll
        for (int i = 0; i < 8; ++i) {
            int   ml   = i + mhalf;
            float gate = 1.f / (1.f + __expf(-c[i]));
            float gt   = Gt[(size_t)sMap[ml] * DDIM + col];
            float gh   = Gh[(size_t)sRow[ml] * DDIM + col];
            sOut[ml * DDIM + col] = gt * gate + gh;
        }
    }
    __syncthreads();

    // ---- phase 4: LayerNorm each staged row, scatter to out[hmap[m]] ----
#pragma unroll
    for (int rr = 0; rr < 2; ++rr) {
        int r  = wave * 2 + rr;
        int cc = lane * 8;
        fx4 x0 = *(const fx4*)(sOut + r * DDIM + cc);
        fx4 x1 = *(const fx4*)(sOut + r * DDIM + cc + 4);
        float s = 0.f, ss = 0.f;
#pragma unroll
        for (int j = 0; j < 4; ++j) { s += x0[j]; ss += x0[j] * x0[j]; }
#pragma unroll
        for (int j = 0; j < 4; ++j) { s += x1[j]; ss += x1[j] * x1[j]; }
#pragma unroll
        for (int o = 16; o > 0; o >>= 1) {
            s  += __shfl_xor(s,  o);
            ss += __shfl_xor(ss, o);
        }
        float mu  = s * (1.f / DDIM);
        float var = ss * (1.f / DDIM) - mu * mu;
        float rs  = rsqrtf(var + EPS);

        fx4 g0 = *(const fx4*)(gamma + cc);
        fx4 g1 = *(const fx4*)(gamma + cc + 4);
        fx4 b0 = *(const fx4*)(beta + cc);
        fx4 b1 = *(const fx4*)(beta + cc + 4);
        fx4 y0, y1;
#pragma unroll
        for (int j = 0; j < 4; ++j) y0[j] = (x0[j] - mu) * rs * g0[j] + b0[j];
#pragma unroll
        for (int j = 0; j < 4; ++j) y1[j] = (x1[j] - mu) * rs * g1[j] + b1[j];

        float* o = out + (size_t)sMap[r] * DDIM + cc;
        *(fx4*)o       = y0;
        *(fx4*)(o + 4) = y1;
    }
}

// ---------------------------------------------------------------------------
// Launch: convert W -> bf16 fragments (ws), base LN over all G_t rows, then
// gate GEMM + update + LN overwriting mapped rows (stream-ordered, unique
// indices => deterministic).
// in: 0=G_t 1=G_h 2=q 3=h_map 4=W_gate 5=ln_gamma 6=ln_beta
// ---------------------------------------------------------------------------
extern "C" void kernel_launch(void* const* d_in, const int* in_sizes, int n_in,
                              void* d_out, int out_size, void* d_ws, size_t ws_size,
                              hipStream_t stream) {
    const float* Gt    = (const float*)d_in[0];
    const float* Gh    = (const float*)d_in[1];
    const float* q     = (const float*)d_in[2];
    const int*   hmap  = (const int*)d_in[3];
    const float* W     = (const float*)d_in[4];
    const float* gamma = (const float*)d_in[5];
    const float* beta  = (const float*)d_in[6];
    float* out = (float*)d_out;
    unsigned short* wbf = (unsigned short*)d_ws;   // 768*256*2 = 393216 B

    int N = in_sizes[0] / DDIM;
    int M = in_sizes[1] / DDIM;

    int convThreads = NTILES * KTILES * 32;                 // 12288
    wconvert_kernel<<<(convThreads + 255) / 256, 256, 0, stream>>>(W, wbf);
    lnall_kernel<<<(N + 7) / 8, 256, 0, stream>>>(Gt, gamma, beta, out, N);
    gate_kernel<<<(M + 15) / 16, 256, 0, stream>>>(Gt, Gh, q, hmap, wbf,
                                                   gamma, beta, out, M);
}
